// SelfAttention_80951543595683
// MI455X (gfx1250) — matrix operand order repaired and checked by reading, not compiled, not run on hardware
//
#include <hip/hip_runtime.h>
#include <hip/hip_bf16.h>
#include <math.h>
#include <stdint.h>

// ---------------------------------------------------------------------------
// Types for CDNA5 WMMA (wave32): V_WMMA_F32_16X16X32_F16
// ---------------------------------------------------------------------------
typedef __attribute__((ext_vector_type(16))) _Float16 v16h;
typedef __attribute__((ext_vector_type(8)))  _Float16 half8;
typedef __attribute__((ext_vector_type(8)))  float    v8f;
typedef __attribute__((ext_vector_type(4)))  unsigned int u32x4;
typedef __attribute__((ext_vector_type(8)))  int      i32x8;
typedef __attribute__((ext_vector_type(4)))  int      i32x4;

static __device__ __forceinline__ v16h cat16(half8 lo, half8 hi) {
  return __builtin_shufflevector(lo, hi, 0,1,2,3,4,5,6,7,8,9,10,11,12,13,14,15);
}

static __device__ __forceinline__ v8f wmma_f16(v16h a, v16h b, v8f c) {
  // (neg_a, A, neg_b, B, c_mod, C, reuse_a, reuse_b)
  return __builtin_amdgcn_wmma_f32_16x16x32_f16(false, a, false, b, (short)0, c,
                                                false, false);
}

// ---------------------------------------------------------------------------
// TDM: async 2-D tile load Global -> LDS (Tensor Data Mover, gfx1250).
// Descriptor per CDNA5 ISA ch.8: group0 = {flags, lds_addr, global_addr, type},
// group1 = {mask/data_size/pad, tensor dims, tile dims, strides}.
// data_size = 2 bytes; pad_enable: insert 32B after every 128B row so the
// staged rows are 16B-aligned and bank-spread (LDS row stride = 160B).
// This toolchain exposes the 6-arg builtin:
//   (u32x4 g0, i32x8 g1, i32x4 g2, i32x4 g3, i32x8 g4, i32 cpol)
// ---------------------------------------------------------------------------
#define KROW 80   // padded LDS row stride in halves (64 data + 16 pad)

static __device__ __forceinline__ void tdm_load_k_tile(uint32_t lds_byte_addr,
                                                       const _Float16* gptr) {
  const uint64_t ga = (uint64_t)(uintptr_t)gptr;
  u32x4 g0;
  g0[0] = 1u;                                           // count=1 (valid), user
  g0[1] = lds_byte_addr;                                // LDS dest (bytes)
  g0[2] = (uint32_t)(ga & 0xffffffffu);                 // global_addr[31:0]
  g0[3] = (uint32_t)((ga >> 32) & 0x01ffffffu) | (2u << 30);  // [56:32] | type=2
  i32x8 g1;
  g1[0] = (int)((1u << 16) |        // data_size = 2 bytes
                (1u << 20) |        // pad_enable
                (4u << 22) |        // pad_interval: 32 DWORDs (128B row)
                (7u << 25));        // pad_amount:   8 DWORDs (32B)
  g1[1] = (int)(64u << 16);                             // tensor_dim0 = 64
  g1[2] = (int)((64u >> 16) | (32u << 16));             // tensor_dim1 = 32
  g1[3] = (int)((32u >> 16) | (64u << 16));             // tile_dim0 = 64
  g1[4] = (int)32u;                                     // tile_dim1 = 32
  g1[5] = (int)512u;                                    // tensor_dim0_stride = A
  g1[6] = 0;
  g1[7] = 0;
  i32x4 z4 = {0, 0, 0, 0};
  i32x8 z8 = {0, 0, 0, 0, 0, 0, 0, 0};
  __builtin_amdgcn_tensor_load_to_lds(g0, g1, z4, z4, z8, 0);
}

// ---------------------------------------------------------------------------
// Problem constants
// ---------------------------------------------------------------------------
#define BB 4
#define SS 2048
#define DD 1024
#define AA 512
#define MTOT (BB * SS)   // 8192

// ---------------------------------------------------------------------------
// Kernel 0a: vectorized fp32 -> f16 conversion (8 elems/thread)
// ---------------------------------------------------------------------------
__global__ void cvt_f32_to_f16_v8(const float* __restrict__ in,
                                  _Float16* __restrict__ out, int n8) {
  int i = blockIdx.x * blockDim.x + threadIdx.x;
  if (i >= n8) return;
  const int base = i * 8;
  float4 a = *(const float4*)(in + base);
  float4 b = *(const float4*)(in + base + 4);
  half8 h;
  h[0] = (_Float16)a.x; h[1] = (_Float16)a.y;
  h[2] = (_Float16)a.z; h[3] = (_Float16)a.w;
  h[4] = (_Float16)b.x; h[5] = (_Float16)b.y;
  h[6] = (_Float16)b.z; h[7] = (_Float16)b.w;
  *(half8*)(out + base) = h;
}

// ---------------------------------------------------------------------------
// Kernel 0b: fused transpose + convert: in[K,N] f32 -> out[N,K] f16
// ---------------------------------------------------------------------------
__global__ void transpose_cvt_f16(const float* __restrict__ in,
                                  _Float16* __restrict__ out, int K, int N) {
  int i = blockIdx.x * blockDim.x + threadIdx.x;
  if (i >= K * N) return;
  const int k = i / N;
  const int n = i - k * N;
  out[(size_t)n * K + k] = (_Float16)in[i];
}

// ---------------------------------------------------------------------------
// Kernel 1: Out = (X[M,K] @ Wt[N,K]^T + bias[N]) * scale,  f16 in/out, f32 acc
// Each wave computes a 16x64 strip: 4 accumulators, A fragment reused 4x.
// All WMMA operands are contiguous 16B vector loads.
// transposeOut=1 stores Out as [N, M] (for V) with a single half8 store.
// ---------------------------------------------------------------------------
__global__ void __launch_bounds__(256)
gemm_bias_wmma4(const _Float16* __restrict__ X, const _Float16* __restrict__ Wt,
                const float* __restrict__ bias, _Float16* __restrict__ Out,
                int M, int N, int K, float scale, int transposeOut) {
  const int wave = threadIdx.x >> 5;
  const int lane = threadIdx.x & 31;
  const int lhi  = (lane < 16) ? 0 : 1;
  const int l15  = lane & 15;

  const int ntn4 = N >> 6;                 // 64-wide strips
  const int wid  = blockIdx.x * 8 + wave;
  if (wid >= (M >> 4) * ntn4) return;
  const int tm = (wid / ntn4) * 16;
  const int tn = (wid % ntn4) * 64;

  v8f acc[4];
#pragma unroll
  for (int t = 0; t < 4; ++t) acc[t] = (v8f){};

  const int arow = tm + l15;

  for (int kb = 0; kb < K; kb += 32) {
    const int a0 = kb + lhi * 16;
    half8 alo = *(const half8*)&X[(size_t)arow * K + a0];
    half8 ahi = *(const half8*)&X[(size_t)arow * K + a0 + 8];
    v16h Af = cat16(alo, ahi);
    const int kk0 = kb + lhi * 16;
#pragma unroll
    for (int t = 0; t < 4; ++t) {
      const int bcol = tn + t * 16 + l15;
      const _Float16* wp = &Wt[(size_t)bcol * K + kk0];
      v16h Bf = cat16(*(const half8*)wp, *(const half8*)(wp + 8));
      acc[t] = wmma_f16(Af, Bf, acc[t]);
    }
  }

#pragma unroll
  for (int t = 0; t < 4; ++t) {
    const int bcol = tn + t * 16 + l15;
    const float bn = bias[bcol];
    if (transposeOut) {
      half8 st;
#pragma unroll
      for (int r = 0; r < 8; ++r) st[r] = (_Float16)((acc[t][r] + bn) * scale);
      *(half8*)&Out[(size_t)bcol * M + tm + lhi * 8] = st;   // contiguous in M
    } else {
#pragma unroll
      for (int r = 0; r < 8; ++r) {
        const int mrow = tm + r + lhi * 8;
        Out[(size_t)mrow * N + bcol] = (_Float16)((acc[t][r] + bn) * scale);
      }
    }
  }
}

// ---------------------------------------------------------------------------
// Kernel 2: fused flash attention.
// Grid: (S/16, B). Block: 256 threads = 8 waves. Block owns 16 query rows.
// Per 32-key step:
//   - TDM (tensor_load_to_lds) streams the NEXT K tile slice per wave into a
//     double-buffered LDS stage; s_wait_tensorcnt(1) guarantees the current
//     one has landed (per-wave in-order completion -> no barriers needed).
//   phase1: waves split A=512 reduction dim (64 each) -> partial 16x32 scores
//   phase2: LDS reduction of 8 partials (serial sum per element)
//   phase3: online softmax (running max / sum), P -> f16 in LDS
//   phase4: each wave: rescale its 128-col O slice, O += P @ V_slice
// V is pre-transposed (Vt[D][B*S]) so phase-4 B fragments are contiguous.
// ---------------------------------------------------------------------------
__global__ void __launch_bounds__(256)
flash_attn_wmma(const _Float16* __restrict__ Qh, const _Float16* __restrict__ Kh,
                const _Float16* __restrict__ Vt, float* __restrict__ Out) {
  __shared__ __align__(16) _Float16 q_s[16 * AA];         // 16 KB
  __shared__ __align__(16) _Float16 k_s[2][8][32 * KROW]; // 80 KB (TDM staged)
  __shared__ float   part[8][16][32];                     // 16 KB
  __shared__ float   s_f[16][32];                         // 2 KB
  __shared__ __align__(16) _Float16 p_s[16 * 32];         // 1 KB
  __shared__ float   m_s[16], l_s[16], alpha_s[16];

  const int b    = blockIdx.y;
  const int q0   = blockIdx.x * 16;
  const int tid  = threadIdx.x;
  const int wave = tid >> 5;
  const int lane = tid & 31;
  const int lhi  = (lane < 16) ? 0 : 1;
  const int l15  = lane & 15;
  const int rowbase = b * SS + q0;
  const int aw   = wave * 64;
  const int NJ   = SS / 32;

  // kick off TDM for the first K tile (per-wave 32x64 slice of K)
  tdm_load_k_tile((uint32_t)(uintptr_t)&k_s[0][wave][0],
                  &Kh[(size_t)(b * SS) * AA + aw]);

  // stage Q tile into LDS (coalesced 16B chunks)
  for (int c = tid; c < (16 * AA) / 8; c += 256) {
    const int r = c >> 6;                 // 64 chunks per 512-half row
    const int ci = (c & 63) * 8;
    *(half8*)&q_s[r * AA + ci] = *(const half8*)&Qh[(size_t)(rowbase + r) * AA + ci];
  }
  if (tid < 16) { m_s[tid] = -3.0e38f; l_s[tid] = 0.0f; }
  __syncthreads();

  v8f o_acc[8];
#pragma unroll
  for (int t = 0; t < 8; ++t) o_acc[t] = (v8f){};

  for (int j = 0; j < NJ; ++j) {
    const int kb = j * 32;

    // ---- issue next K tile via TDM; wait for current one ----
    if (j + 1 < NJ) {
      tdm_load_k_tile((uint32_t)(uintptr_t)&k_s[(j + 1) & 1][wave][0],
                      &Kh[(size_t)(b * SS + kb + 32) * AA + aw]);
      __builtin_amdgcn_s_wait_tensorcnt(1);
      // prefetch next V slice into cache while we compute
      __builtin_prefetch(&Vt[(size_t)(wave * 128 + lane * 4) * MTOT + b * SS + kb + 32],
                         0, 1);
    } else {
      __builtin_amdgcn_s_wait_tensorcnt(0);
    }
    const _Float16* kt = &k_s[j & 1][wave][0];

    // ---- phase 1: partial scores for this wave's A-slice (64 wide) ----
    v16h Aq[2];
#pragma unroll
    for (int ks = 0; ks < 2; ++ks) {
      const int a0 = aw + ks * 32 + lhi * 16;
      half8 alo = *(const half8*)&q_s[l15 * AA + a0];
      half8 ahi = *(const half8*)&q_s[l15 * AA + a0 + 8];
      Aq[ks] = cat16(alo, ahi);
    }
    v8f sacc[2];
#pragma unroll
    for (int nh = 0; nh < 2; ++nh) {
      v8f acc = {};
#pragma unroll
      for (int ks = 0; ks < 2; ++ks) {
        const _Float16* kp = kt + (nh * 16 + l15) * KROW + ks * 32 + lhi * 16;
        v16h Bf = cat16(*(const half8*)kp, *(const half8*)(kp + 8));
        acc = wmma_f16(Aq[ks], Bf, acc);
      }
      sacc[nh] = acc;
    }
#pragma unroll
    for (int nh = 0; nh < 2; ++nh)
#pragma unroll
      for (int r = 0; r < 8; ++r)
        part[wave][r + lhi * 8][nh * 16 + l15] = sacc[nh][r];
    __syncthreads();

    // ---- phase 2: reduce 8 partials ----
    for (int e = tid; e < 512; e += 256) {
      const int mr = e >> 5, nc = e & 31;
      float s = 0.0f;
#pragma unroll
      for (int w = 0; w < 8; ++w) s += part[w][mr][nc];
      s_f[mr][nc] = s;
    }
    __syncthreads();

    // ---- phase 3: online softmax (16 threads, one row each) ----
    if (tid < 16) {
      const float mprev = m_s[tid];
      float mx = mprev;
#pragma unroll
      for (int c = 0; c < 32; ++c) mx = fmaxf(mx, s_f[tid][c]);
      const float al = __expf(mprev - mx);
      float rs = 0.0f;
#pragma unroll
      for (int c = 0; c < 32; ++c) {
        const float p = __expf(s_f[tid][c] - mx);
        p_s[tid * 32 + c] = (_Float16)p;
        rs += p;
      }
      l_s[tid] = l_s[tid] * al + rs;
      m_s[tid] = mx;
      alpha_s[tid] = al;
    }
    __syncthreads();

    // ---- phase 4: rescale O, O += P @ V_slice (Vt rows are contiguous) ----
    float alr[8];
#pragma unroll
    for (int r = 0; r < 8; ++r) alr[r] = alpha_s[r + lhi * 8];
#pragma unroll
    for (int t = 0; t < 8; ++t)
#pragma unroll
      for (int r = 0; r < 8; ++r) o_acc[t][r] *= alr[r];

    const int p0 = lhi * 16;
    half8 plo = *(const half8*)&p_s[l15 * 32 + p0];
    half8 phi = *(const half8*)&p_s[l15 * 32 + p0 + 8];
    v16h Pf = cat16(plo, phi);

#pragma unroll
    for (int t = 0; t < 8; ++t) {
      const int col = wave * 128 + t * 16 + l15;
      const _Float16* vp = &Vt[(size_t)col * MTOT + b * SS + kb + lhi * 16];
      v16h Vf = cat16(*(const half8*)vp, *(const half8*)(vp + 8));
      o_acc[t] = wmma_f16(Pf, Vf, o_acc[t]);
    }
    __syncthreads();
  }

  // epilogue: normalize by running sum, store fp32
#pragma unroll
  for (int t = 0; t < 8; ++t) {
#pragma unroll
    for (int r = 0; r < 8; ++r) {
      const int mrow = r + lhi * 8;
      const float inv = 1.0f / l_s[mrow];
      Out[(size_t)(b * SS + q0 + mrow) * DD + wave * 128 + t * 16 + l15] =
          o_acc[t][r] * inv;
    }
  }
}

// ---------------------------------------------------------------------------
// Host launch
// ---------------------------------------------------------------------------
extern "C" void kernel_launch(void* const* d_in, const int* in_sizes, int n_in,
                              void* d_out, int out_size, void* d_ws, size_t ws_size,
                              hipStream_t stream) {
  const float* x  = (const float*)d_in[0];   // [B,S,D]
  const float* Wq = (const float*)d_in[1];   // [D,A]
  const float* bq = (const float*)d_in[2];   // [A]
  const float* Wk = (const float*)d_in[3];   // [D,A]
  const float* bk = (const float*)d_in[4];   // [A]
  const float* Wv = (const float*)d_in[5];   // [D,D]
  const float* bv = (const float*)d_in[6];   // [D]
  float* out = (float*)d_out;

  char* ws = (char*)d_ws;
  _Float16* Xh  = (_Float16*)(ws);                 // 16 MB  [B*S, D]
  _Float16* Qh  = (_Float16*)(ws + (16u << 20));   //  8 MB  [B*S, A]
  _Float16* Kh  = (_Float16*)(ws + (24u << 20));   //  8 MB  [B*S, A]
  _Float16* Vt  = (_Float16*)(ws + (32u << 20));   // 16 MB  [D, B*S] (transposed)
  _Float16* Wqt = (_Float16*)(ws + (48u << 20));   //  1 MB  [A, D] (transposed)
  _Float16* Wkt = (_Float16*)(ws + (49u << 20));   //  1 MB  [A, D] (transposed)
  _Float16* Wvt = (_Float16*)(ws + (50u << 20));   //  2 MB  [D, D] (transposed)

  const int nX  = BB * SS * DD;   // 8388608
  const int nWa = DD * AA;        // 524288
  const int nWv = DD * DD;        // 1048576

  // 0) convert X (vectorized), transpose+convert weights
  cvt_f32_to_f16_v8<<<(nX / 8 + 255) / 256, 256, 0, stream>>>(x, Xh, nX / 8);
  transpose_cvt_f16<<<(nWa + 255) / 256, 256, 0, stream>>>(Wq, Wqt, DD, AA);
  transpose_cvt_f16<<<(nWa + 255) / 256, 256, 0, stream>>>(Wk, Wkt, DD, AA);
  transpose_cvt_f16<<<(nWv + 255) / 256, 256, 0, stream>>>(Wv, Wvt, DD, DD);

  // 1) projections (scale 1/sqrt(A) folded into Q; V stored transposed)
  const float qscale = 0.044194173824159216f;  // 1/sqrt(512)
  {
    const int waves = (MTOT / 16) * (AA / 64);   // 4096
    gemm_bias_wmma4<<<waves / 8, 256, 0, stream>>>(Xh, Wqt, bq, Qh, MTOT, AA, DD,
                                                   qscale, 0);
    gemm_bias_wmma4<<<waves / 8, 256, 0, stream>>>(Xh, Wkt, bk, Kh, MTOT, AA, DD,
                                                   1.0f, 0);
  }
  {
    const int waves = (MTOT / 16) * (DD / 64);   // 8192
    gemm_bias_wmma4<<<waves / 8, 256, 0, stream>>>(Xh, Wvt, bv, Vt, MTOT, DD, DD,
                                                   1.0f, 1);
  }

  // 2) fused flash attention
  flash_attn_wmma<<<dim3(SS / 16, BB), 256, 0, stream>>>(Qh, Kh, Vt, out);
}
